// EntropicOT_13314398618445
// MI455X (gfx1250) — matrix operand order repaired and checked by reading, not compiled, not run on hardware
//
#include <hip/hip_runtime.h>
#include <hip/hip_bf16.h>

// ---------------------------------------------------------------------------
// Entropic OT / Sinkhorn on MI455X (gfx1250, wave32)
//   K, K^T materialized in bf16 (67 MB total -> L2 resident on 192MB L2)
//   GEMVs done as broadcast-WMMA: v_wmma_f32_16x16x32_bf16, fp32 accumulate
//   Inner loop is software-pipelined (register ping-pong); a laundered zero
//   offset (asm volatile) blocks LICM from hoisting the iteration-invariant
//   K loads out of the Sinkhorn loop (previously caused scratch spills),
//   while keeping the base pointer's global address-space provenance so the
//   loads stay global_load_b128 (not flat_load, which would contend with the
//   ds_load B-operand stream on the LDS pipe and bump both counters).
// ---------------------------------------------------------------------------

#define OT_B 64
#define OT_N 512
#define OT_M 512
#define OT_ITERS 50
#define OT_INVEPS 20.0f      // 1/0.05
#define OT_STAB 1e-8f
#define OT_AB (1.0f / 512.0f)

typedef __attribute__((ext_vector_type(16))) __bf16 v16bf;
typedef __attribute__((ext_vector_type(8)))  float  v8f;

union OTTile32B {            // 32 bytes: one WMMA bf16 operand per lane
    v16bf v;
    uint4 q[2];
};

// ---------------------------------------------------------------------------
// Kernel 1: Kh = bf16(exp(-C/eps)) row-major, and KTh = transpose(Kh)
// 32x32 tiles, 32x8 threads, LDS transpose for coalesced KT writes.
// ---------------------------------------------------------------------------
__global__ __launch_bounds__(256)
void EntropicOT_exp_transpose(const float* __restrict__ C,
                              __bf16* __restrict__ Kh,
                              __bf16* __restrict__ KTh) {
    __shared__ __bf16 tile[32][33];
    const int b  = blockIdx.z;
    const int m0 = blockIdx.x * 32;
    const int n0 = blockIdx.y * 32;
    const float* Cb   = C   + (size_t)b * OT_N * OT_M;
    __bf16*      Khb  = Kh  + (size_t)b * OT_N * OT_M;
    __bf16*      KThb = KTh + (size_t)b * OT_N * OT_M;

    const int tx = threadIdx.x;   // 0..31
    const int ty = threadIdx.y;   // 0..7

#pragma unroll
    for (int r = ty; r < 32; r += 8) {
        const int n = n0 + r;
        const int m = m0 + tx;
        const float c = Cb[(size_t)n * OT_M + m];
        const __bf16 kh = (__bf16)__expf(-OT_INVEPS * c);
        Khb[(size_t)n * OT_M + m] = kh;    // coalesced row-major K
        tile[r][tx] = kh;
    }
    __syncthreads();
#pragma unroll
    for (int r = ty; r < 32; r += 8) {
        const int m = m0 + r;
        const int n = n0 + tx;
        KThb[(size_t)m * OT_N + n] = tile[tx][r];  // coalesced row-major K^T
    }
}

// ---------------------------------------------------------------------------
// Broadcast-GEMV via WMMA: y[m0 + 0..15] = sum_k Mtx[m0+i][k] * x[k]
//   A (16x32 bf16) : rows of Mtx (row-major), per-ISA A operand layout
//   B (32x16 bf16) : x replicated across the 16 columns (from LDS)
//   D (16x16 f32)  : every column holds the same 16 partial sums
// Double-buffered: chunk i+1 loads are in flight while WMMA consumes chunk i.
// Runs with full EXEC (no divergence around the WMMA) as the ISA requires.
// ---------------------------------------------------------------------------
__device__ __forceinline__
void ot_wave_gemv(const __bf16* Mtx,                // [512][512] row-major
                  const __bf16* __restrict__ xh,    // LDS bf16 vector [512]
                  float* __restrict__ y,            // LDS f32 output [512]
                  int m0, int lane) {
    // Laundered zero offset: defeats LICM across the 50-iteration Sinkhorn
    // loop (no spill-to-scratch of the whole K stream) while preserving the
    // kernel-arg provenance of Mtx so loads lower to global_load_b128.
    unsigned long long zofs = 0;
    asm volatile("" : "+v"(zofs));
    Mtx += zofs;

    const int rowSel = lane & 15;
    const int hi     = lane >> 4;                   // 0 or 1
    // Per-lane A base: row (m0+rowSel); lanes 0-15 take K=kb..kb+7 / kb+16..+23,
    // lanes 16-31 take K=kb+8..+15 / kb+24..+31 (16-bit A operand layout).
    const __bf16* Arow = Mtx + (size_t)(m0 + rowSel) * OT_M + hi * 8;
    // Per-lane B base: lanes 0-15 take x[kb..kb+7] / x[kb+16..+23], lanes 16-31 x[kb+8..+15] / x[kb+24..+31]
    const __bf16* Bvec = xh + hi * 8;

    v8f acc = {};
    OTTile32B a[2], bb[2];

    // Prologue: chunk 0 in flight.
    a[0].q[0]  = *(const uint4*)(Arow);
    a[0].q[1]  = *(const uint4*)(Arow + 16);
    bb[0].q[0] = *(const uint4*)(Bvec);
    bb[0].q[1] = *(const uint4*)(Bvec + 16);

#pragma unroll
    for (int i = 0; i < OT_M / 32; ++i) {
        const int cur = i & 1;
        const int nxt = cur ^ 1;
        const int kb1 = (i + 1) * 32;
        if (i + 1 < OT_M / 32) {
            // Issue next chunk's loads before consuming the current one.
            a[nxt].q[0]  = *(const uint4*)(Arow + kb1);
            a[nxt].q[1]  = *(const uint4*)(Arow + kb1 + 16);
            bb[nxt].q[0] = *(const uint4*)(Bvec + kb1);
            bb[nxt].q[1] = *(const uint4*)(Bvec + kb1 + 16);
        }
        acc = __builtin_amdgcn_wmma_f32_16x16x32_bf16(
                  false, a[cur].v, false, bb[cur].v, (short)0, acc,
                  false, false);
    }
    // C/D layout: VGPR r -> M=r (lanes 0-15) / M=8+r (lanes 16-31); N=lane%16
    if ((lane & 15) == 0) {
#pragma unroll
        for (int r = 0; r < 8; ++r)
            y[m0 + hi * 8 + r] = acc[r];
    }
}

// ---------------------------------------------------------------------------
// Kernel 2: fused Sinkhorn iterations + P/loss output. One block per batch.
// 1024 threads = 32 waves; each wave owns one 16-wide output tile.
// ---------------------------------------------------------------------------
__global__ __launch_bounds__(1024)
void EntropicOT_sinkhorn(const float* __restrict__ C,
                         const __bf16* __restrict__ Kh,
                         const __bf16* __restrict__ KTh,
                         float* __restrict__ P,
                         float* __restrict__ loss) {
    __shared__ __attribute__((aligned(16))) float  u[OT_N];
    __shared__ __attribute__((aligned(16))) float  v[OT_M];
    __shared__ __attribute__((aligned(16))) __bf16 uh[OT_N];
    __shared__ __attribute__((aligned(16))) __bf16 vh[OT_M];
    __shared__ __attribute__((aligned(16))) float  y[OT_N];
    __shared__ float red[32];

    const int b = blockIdx.x;
    const __bf16* Kb  = Kh  + (size_t)b * OT_N * OT_M;
    const __bf16* KTb = KTh + (size_t)b * OT_N * OT_M;

    const int t    = threadIdx.x;
    const int wave = t >> 5;
    const int lane = t & 31;
    const int m0   = wave * 16;      // 32 waves x 16 outputs = 512

    if (t < OT_N) {
        u[t]  = OT_AB;
        uh[t] = (__bf16)OT_AB;
    }
    __syncthreads();

    for (int it = 0; it < OT_ITERS; ++it) {
        // y = K^T u   (rows of KTh are columns of K -> contiguous)
        ot_wave_gemv(KTb, uh, y, m0, lane);
        __syncthreads();
        if (t < OT_M) {
            const float vv = OT_AB / (y[t] + OT_STAB);
            v[t]  = vv;
            vh[t] = (__bf16)vv;
        }
        __syncthreads();
        // y = K v
        ot_wave_gemv(Kb, vh, y, m0, lane);
        __syncthreads();
        if (t < OT_N) {
            const float uu = OT_AB / (y[t] + OT_STAB);
            u[t]  = uu;
            uh[t] = (__bf16)uu;
        }
        __syncthreads();
    }

    // P = u (.) exp(-C/eps) (.) v  in fp32 (recompute K at full precision),
    // ot_loss[b] = sum(P * C)
    const float* Cb = C + (size_t)b * OT_N * OT_M;
    float*       Pb = P + (size_t)b * OT_N * OT_M;
    float acc = 0.0f;
    for (int idx = t; idx < OT_N * OT_M; idx += 1024) {
        const int n = idx >> 9;
        const int m = idx & 511;
        const float c  = Cb[idx];
        const float kf = __expf(-OT_INVEPS * c);
        const float p  = u[n] * kf * v[m];
        Pb[idx] = p;
        acc += p * c;
    }
    // wave32 reduction then cross-wave via LDS
#pragma unroll
    for (int off = 16; off > 0; off >>= 1)
        acc += __shfl_down(acc, off);
    if (lane == 0) red[wave] = acc;
    __syncthreads();
    if (wave == 0) {
        float s = red[lane];
#pragma unroll
        for (int off = 16; off > 0; off >>= 1)
            s += __shfl_down(s, off);
        if (lane == 0) loss[b] = s;
    }
}

// ---------------------------------------------------------------------------
// Host-side launcher
// ---------------------------------------------------------------------------
extern "C" void kernel_launch(void* const* d_in, const int* in_sizes, int n_in,
                              void* d_out, int out_size, void* d_ws, size_t ws_size,
                              hipStream_t stream) {
    (void)in_sizes; (void)n_in; (void)out_size; (void)ws_size;

    const float* C = (const float*)d_in[0];

    // d_out: P [64*512*512] fp32, then ot_loss [64] fp32
    float* P    = (float*)d_out;
    float* loss = (float*)d_out + (size_t)OT_B * OT_N * OT_M;

    // Workspace: Kh (33.5 MB bf16) then KTh (33.5 MB bf16) — L2 resident
    __bf16* Kh  = (__bf16*)d_ws;
    __bf16* KTh = Kh + (size_t)OT_B * OT_N * OT_M;

    dim3 g1(OT_M / 32, OT_N / 32, OT_B);
    dim3 b1(32, 8, 1);
    EntropicOT_exp_transpose<<<g1, b1, 0, stream>>>(C, Kh, KTh);

    EntropicOT_sinkhorn<<<dim3(OT_B), dim3(1024), 0, stream>>>(C, Kh, KTh, P, loss);
}
